// MonolithicDecoderLayer_47485158424661
// MI455X (gfx1250) — matrix operand order repaired and checked by reading, not compiled, not run on hardware
//
#include <hip/hip_runtime.h>
#include <hip/hip_bf16.h>
#include <cstdint>
#include <cstddef>

// ---------------------------------------------------------------------------
// MLA decoder layer for MI455X (gfx1250, wave32, WMMA + async-LDS staging).
// All heavy matmuls run through v_wmma_f32_16x16x32_bf16 (f32 accumulate).
// GEMM tile staging uses GLOBAL_LOAD_ASYNC_TO_LDS_B128 (ASYNCcnt-tracked).
// ---------------------------------------------------------------------------

typedef __attribute__((ext_vector_type(16))) __bf16 v16bf;
typedef __attribute__((ext_vector_type(8)))  float  v8f;

#define TT    1024      // tokens
#define HID   2048
#define NEGF  (-1e30f)

union FragBF { v16bf v; uint4 u[2]; };
union Pack8  { uint4 u; __bf16 h[8]; };

#define LDS_AS __attribute__((address_space(3)))

// Raw LDS byte offset of a shared-memory pointer (AS3 pointers are offsets
// from the wave's LDS base, which is what the async-load VDST operand wants).
__device__ inline uint32_t lds_off(const void* p)
{
    return (uint32_t)(uintptr_t)(LDS_AS const void*)p;
}

// Async copy of 16 bytes global -> LDS (no VGPR data path, ASYNCcnt-tracked).
__device__ inline void async_copy_b128(uint32_t lds_byte_off, const void* gptr)
{
    asm volatile("global_load_async_to_lds_b128 %0, %1, off"
                 :: "v"(lds_byte_off), "v"(gptr) : "memory");
}

__device__ inline void async_wait0()
{
    asm volatile("s_wait_asynccnt 0x0" ::: "memory");
}

// ---------------------------------------------------------------------------
// Generic tiled WMMA GEMM:  C[M,N] (+)= A[M,K] * op(B)
//   BT_MODE=true : B is [N,K] row-major (x @ W.T pattern)
//   BT_MODE=false: B is [K,N] row-major (x @ W pattern)
// OUT_MODE: 0 = f32 store, 1 = f32 accumulate, 2 = bf16 store
// EPI:      0 = none, 1 = attention epilogue (scale + causal + topk-threshold)
// Block: 128 threads (4 waves). Tile 64x64, BK=32. M must be multiple of 64.
// ---------------------------------------------------------------------------
template<bool BT_MODE, int OUT_MODE, int EPI>
__global__ __launch_bounds__(128)
void gemm_wmma_k(const __bf16* __restrict__ A, int lda,
                 const __bf16* __restrict__ B, int ldb,
                 void* __restrict__ C, int ldc,
                 int N, int K,
                 float scale,
                 const float* __restrict__ isc,
                 const float* __restrict__ thr)
{
    __shared__ __bf16 As[64][40];   // [m][k], stride 40 halves (80B, 16B-aligned)
    __shared__ __bf16 Bs[32][72];   // [k][n], stride 72 halves (144B, 16B-aligned)

    const int tid  = threadIdx.x;
    const int wid  = tid >> 5;
    const int lane = tid & 31;
    const int hi   = lane >> 4;
    const int m0   = blockIdx.y * 64;
    const int n0   = blockIdx.x * 64;

    v8f acc[4] = {};

    for (int k0 = 0; k0 < K; k0 += 32) {
        __syncthreads();
        // ---- stage A tile (64x32) via async global->LDS b128 copies ----
        for (int c = tid; c < 256; c += 128) {
            int r  = c >> 2;
            int kc = (c & 3) << 3;
            async_copy_b128(lds_off(&As[r][kc]),
                            A + (size_t)(m0 + r) * lda + k0 + kc);
        }
        // ---- stage B tile into [k][n] layout ----
        if (BT_MODE) {
            for (int c = tid; c < 256; c += 128) {
                int n  = c >> 2;
                int kc = (c & 3) << 3;
                Pack8 tmp;
                if (n0 + n < N) {
                    const __bf16* gp = B + (size_t)(n0 + n) * ldb + k0 + kc;
                    tmp.u = *(const uint4*)gp;
                    if (k0 + 32 < K) __builtin_prefetch(gp + 32, 0, 0);
                } else {
                    tmp.u = make_uint4(0u, 0u, 0u, 0u);
                }
#pragma unroll
                for (int j = 0; j < 8; ++j) Bs[kc + j][n] = tmp.h[j];
            }
        } else {
            for (int c = tid; c < 256; c += 128) {
                int k  = c >> 3;
                int nc = (c & 7) << 3;
                if (n0 + nc + 8 <= N) {
                    async_copy_b128(lds_off(&Bs[k][nc]),
                                    B + (size_t)(k0 + k) * ldb + n0 + nc);
                } else {
#pragma unroll
                    for (int j = 0; j < 8; ++j)
                        Bs[k][nc + j] = (n0 + nc + j < N)
                            ? B[(size_t)(k0 + k) * ldb + n0 + nc + j]
                            : (__bf16)0.0f;
                }
            }
        }
        async_wait0();          // all of this wave's async LDS writes landed
        __syncthreads();        // cross-wave visibility

        // ---- fragments per documented wave32 layouts ----
        FragBF af;
        const int row = wid * 16 + (lane & 15);
        af.u[0] = *(const uint4*)&As[row][8 * hi];        // K j+8*hi      (j=0..7)
        af.u[1] = *(const uint4*)&As[row][16 + 8 * hi];   // K j+8+8*hi    (j=8..15)
#pragma unroll
        for (int nt = 0; nt < 4; ++nt) {
            FragBF bfm;                                    // lane%16 = N, elements = K (table order)
#pragma unroll
            for (int s = 0; s < 16; ++s)
                bfm.v[s] = Bs[8 * hi + (s & 7) + 16 * (s >> 3)][nt * 16 + (lane & 15)];
            acc[nt] = __builtin_amdgcn_wmma_f32_16x16x32_bf16(
                false, af.v, false, bfm.v, (short)0, acc[nt], false, false);
        }
    }

    // ---- epilogue: C/D layout lane<16 -> N=lane, M=g ; lane>=16 -> M=g+8 ----
#pragma unroll
    for (int nt = 0; nt < 4; ++nt) {
        int ncol = n0 + nt * 16 + (lane & 15);
        if (ncol >= N) continue;
#pragma unroll
        for (int g = 0; g < 8; ++g) {
            int r = m0 + wid * 16 + g + 8 * hi;
            float v = acc[nt][g];
            if (EPI == 1) {
                v *= scale;
                bool ok = (ncol <= r) && (isc[(size_t)r * TT + ncol] >= thr[r]);
                if (!ok) v = NEGF;
            }
            size_t off = (size_t)r * ldc + ncol;
            if (OUT_MODE == 2)      ((__bf16*)C)[off] = (__bf16)v;
            else if (OUT_MODE == 1) ((float*)C)[off] += v;
            else                    ((float*)C)[off]  = v;
        }
    }
}

template<bool BT, int OUT, int EPI>
static void gemm(hipStream_t s, const __bf16* A, int lda, const __bf16* B, int ldb,
                 void* C, int ldc, int M, int N, int K,
                 float scale = 1.0f, const float* isc = nullptr,
                 const float* thr = nullptr)
{
    dim3 grid((N + 63) / 64, M / 64);
    gemm_wmma_k<BT, OUT, EPI><<<grid, 128, 0, s>>>(A, lda, B, ldb, C, ldc,
                                                   N, K, scale, isc, thr);
}

// ---------------------------------------------------------------------------
// f32 -> bf16 convert (weights / misc)
// ---------------------------------------------------------------------------
__global__ void cvt_bf16_k(const float* __restrict__ in, __bf16* __restrict__ out,
                           int n)
{
    for (int i = blockIdx.x * blockDim.x + threadIdx.x; i < n;
         i += gridDim.x * blockDim.x)
        out[i] = (__bf16)in[i];
}

// ---------------------------------------------------------------------------
// helpers
// ---------------------------------------------------------------------------
__device__ inline float block_reduce_sum256(float v, float* sbuf)
{
    int tid = threadIdx.x;
    sbuf[tid] = v; __syncthreads();
    for (int s = 128; s > 0; s >>= 1) {
        if (tid < s) sbuf[tid] += sbuf[tid + s];
        __syncthreads();
    }
    float r = sbuf[0]; __syncthreads();
    return r;
}

__device__ inline void rope_cs(float pos, int pair, float& c, float& s)
{
    float f = pos * __powf(10000.0f, -(float)pair * (1.0f / 32.0f));
    c = __cosf(f); s = __sinf(f);
}

// ---------------------------------------------------------------------------
// res = a + b ; h = rms(res)*w  -> bf16 (+ optional f32 copy)
// ---------------------------------------------------------------------------
__global__ void resrms_k(const float* __restrict__ a, const float* __restrict__ b,
                         const float* __restrict__ w,
                         float* __restrict__ res_out,
                         __bf16* __restrict__ h_bf,
                         float* __restrict__ h_f32)
{
    __shared__ float sbuf[256];
    int t = blockIdx.x;
    float ss = 0.0f;
    for (int i = threadIdx.x; i < HID; i += 256) {
        float v = a[(size_t)t * HID + i] + b[(size_t)t * HID + i];
        res_out[(size_t)t * HID + i] = v;
        ss += v * v;
    }
    ss = block_reduce_sum256(ss, sbuf);
    float r = rsqrtf(ss * (1.0f / HID) + 1e-6f);
    for (int i = threadIdx.x; i < HID; i += 256) {
        float v = res_out[(size_t)t * HID + i] * r * w[i];
        h_bf[(size_t)t * HID + i] = (__bf16)v;
        if (h_f32) h_f32[(size_t)t * HID + i] = v;
    }
}

// ---------------------------------------------------------------------------
// post-s1: q_c rms -> bf16 ; kv_c rms + roped k_pe -> kv_bf[T][576];
// idx_k LN + rope -> bf16 ; idx_w scale -> f32
// s1 row layout: [0,768) q_c | [768,1280) kv_c | [1280,1344) k_pe |
//                [1344,1472) idx_k | [1472,1504) idx_w
// ---------------------------------------------------------------------------
__global__ void prep1_k(const float* __restrict__ s1, const int* __restrict__ pos,
                        const float* __restrict__ wq, const float* __restrict__ wkv,
                        const float* __restrict__ ikw, const float* __restrict__ ikb,
                        __bf16* __restrict__ qc_bf, __bf16* __restrict__ kv_bf,
                        __bf16* __restrict__ idxk_bf, float* __restrict__ idxw)
{
    __shared__ float sbuf[256];
    __shared__ float L[128];
    int t = blockIdx.x, tid = threadIdx.x;
    const float* row = s1 + (size_t)t * 1504;
    float pt = (float)pos[t];

    // q_c rms (768)
    float ss = 0.0f;
    for (int i = tid; i < 768; i += 256) { float v = row[i]; ss += v * v; }
    ss = block_reduce_sum256(ss, sbuf);
    float r = rsqrtf(ss * (1.0f / 768.0f) + 1e-6f);
    for (int i = tid; i < 768; i += 256)
        qc_bf[(size_t)t * 768 + i] = (__bf16)(row[i] * r * wq[i]);

    // kv_c rms (512)
    ss = 0.0f;
    for (int i = tid; i < 512; i += 256) { float v = row[768 + i]; ss += v * v; }
    ss = block_reduce_sum256(ss, sbuf);
    r = rsqrtf(ss * (1.0f / 512.0f) + 1e-6f);
    for (int i = tid; i < 512; i += 256)
        kv_bf[(size_t)t * 576 + i] = (__bf16)(row[768 + i] * r * wkv[i]);

    // k_pe rope (64) -> kv_bf cols 512..575
    if (tid < 64) {
        int j = tid, p = j & 31;
        float c, s; rope_cs(pt, p, c, s);
        float v = (j < 32) ? (row[1280 + j] * c - row[1280 + j + 32] * s)
                           : (row[1280 + j] * c + row[1280 + j - 32] * s);
        kv_bf[(size_t)t * 576 + 512 + j] = (__bf16)v;
    }

    // idx_k LN (128) + rope first 64
    float sm = 0.0f;
    for (int i = tid; i < 128; i += 256) sm += row[1344 + i];
    sm = block_reduce_sum256(sm, sbuf) * (1.0f / 128.0f);
    float sv = 0.0f;
    for (int i = tid; i < 128; i += 256) { float d = row[1344 + i] - sm; sv += d * d; }
    sv = block_reduce_sum256(sv, sbuf) * (1.0f / 128.0f);
    float rv = rsqrtf(sv + 1e-6f);
    if (tid < 128) L[tid] = (row[1344 + tid] - sm) * rv * ikw[tid] + ikb[tid];
    __syncthreads();
    if (tid < 128) {
        int j = tid; float v;
        if (j < 64) {
            int p = j & 31; float c, s; rope_cs(pt, p, c, s);
            v = (j < 32) ? (L[j] * c - L[j + 32] * s) : (L[j] * c + L[j - 32] * s);
        } else v = L[j];
        idxk_bf[(size_t)t * 128 + j] = (__bf16)v;
    }
    if (tid < 32)
        idxw[t * 32 + tid] = row[1472 + tid] * 0.17677669529663687f; // 32^-0.5
}

// ---------------------------------------------------------------------------
// post-s3: idx_q rope+scale -> bf16[T][32*128]; q_nope -> bf16[T][16*128];
// q_pe rope -> mqa_q_bf[h][T][576] cols 512..575
// s3 row: [0,4096) idx_q | [4096,7168) q (16 heads x 192)
// ---------------------------------------------------------------------------
__global__ void prep2_k(const float* __restrict__ s3, const int* __restrict__ pos,
                        __bf16* __restrict__ idxq_bf,
                        __bf16* __restrict__ qnope_bf,
                        __bf16* __restrict__ mqa_bf)
{
    int t = blockIdx.x, tid = threadIdx.x;
    const float* row = s3 + (size_t)t * 7168;
    float pt = (float)pos[t];
    const float iscl = 0.08838834764831845f; // 128^-0.5

    for (int j = tid; j < 4096; j += 256) {
        int d = j & 127;
        float v;
        if (d < 64) {
            int p = d & 31; float c, s; rope_cs(pt, p, c, s);
            v = (d < 32) ? (row[j] * c - row[j + 32] * s)
                         : (row[j] * c + row[j - 32] * s);
        } else v = row[j];
        idxq_bf[(size_t)t * 4096 + j] = (__bf16)(v * iscl);
    }
    for (int j = tid; j < 16 * 192; j += 256) {
        int h = j / 192, d = j % 192;
        const float* q = row + 4096 + h * 192;
        if (d < 128) {
            qnope_bf[(size_t)t * 2048 + h * 128 + d] = (__bf16)q[d];
        } else {
            int i = d - 128, p = i & 31;
            float c, s; rope_cs(pt, p, c, s);
            float v = (i < 32) ? (q[128 + i] * c - q[128 + i + 32] * s)
                               : (q[128 + i] * c + q[128 + i - 32] * s);
            mqa_bf[(size_t)h * TT * 576 + (size_t)t * 576 + 512 + i] = (__bf16)v;
        }
    }
}

// ---------------------------------------------------------------------------
// Indexer scores (WMMA): per block (t, s-tile of 64):
//   S = idx_q[t] (32x128) @ idx_k[s0:s0+64]^T, iscores = sum_h relu(S)*w[t,h]
// 2 waves; wave w handles heads [16w,16w+16).
// ---------------------------------------------------------------------------
__global__ __launch_bounds__(64)
void idx_score_k(const __bf16* __restrict__ idxq, const __bf16* __restrict__ idxk,
                 const float* __restrict__ idxw, float* __restrict__ iscores)
{
    __shared__ __bf16 Aq[32][136];    // heads x d
    __shared__ __bf16 Bk[128][72];    // d x s
    __shared__ float  red[2][64];

    int t = blockIdx.x, n0 = blockIdx.y * 64;
    int tid = threadIdx.x, wid = tid >> 5, lane = tid & 31, hi = lane >> 4;

    for (int c = tid; c < 512; c += 64) {          // 32*128 halves / 8, async
        int h = c >> 4, dc = (c & 15) << 3;
        async_copy_b128(lds_off(&Aq[h][dc]),
                        idxq + (size_t)t * 4096 + h * 128 + dc);
    }
    for (int c = tid; c < 1024; c += 64) {         // 64*128 halves / 8
        int s = c >> 4, dc = (c & 15) << 3;
        Pack8 tmp;
        tmp.u = *(const uint4*)(idxk + (size_t)(n0 + s) * 128 + dc);
#pragma unroll
        for (int j = 0; j < 8; ++j) Bk[dc + j][s] = tmp.h[j];
    }
    async_wait0();
    __syncthreads();

    v8f acc[4] = {};
    const int row = wid * 16 + (lane & 15);
#pragma unroll
    for (int kt = 0; kt < 4; ++kt) {
        FragBF af;
        af.u[0] = *(const uint4*)&Aq[row][kt * 32 + 8 * hi];
        af.u[1] = *(const uint4*)&Aq[row][kt * 32 + 16 + 8 * hi];
#pragma unroll
        for (int nt = 0; nt < 4; ++nt) {
            FragBF bfm;
#pragma unroll
            for (int s = 0; s < 16; ++s)
                bfm.v[s] = Bk[kt * 32 + 8 * hi + (s & 7) + 16 * (s >> 3)][nt * 16 + (lane & 15)];
            acc[nt] = __builtin_amdgcn_wmma_f32_16x16x32_bf16(
                false, af.v, false, bfm.v, (short)0, acc[nt], false, false);
        }
    }

    // relu * head-weight, reduce over the 8 heads this lane holds, then xor-16
#pragma unroll
    for (int nt = 0; nt < 4; ++nt) {
        float p = 0.0f;
#pragma unroll
        for (int g = 0; g < 8; ++g) {
            int head = wid * 16 + g + 8 * hi;
            p += fmaxf(acc[nt][g], 0.0f) * idxw[t * 32 + head];
        }
        p += __shfl_xor(p, 16, 32);
        if (lane < 16) red[wid][nt * 16 + lane] = p;
    }
    __syncthreads();
    if (tid < 64) {
        int s = n0 + tid;
        float v = red[0][tid] + red[1][tid];
        iscores[(size_t)t * TT + s] = (s <= t) ? v : NEGF;
    }
}

// ---------------------------------------------------------------------------
// Per-row bitonic sort of 1024 scores -> threshold of top-512 selection.
// ---------------------------------------------------------------------------
__global__ __launch_bounds__(512)
void topk_thresh_k(const float* __restrict__ iscores, float* __restrict__ thresh)
{
    __shared__ float s[1024];
    int t = blockIdx.x, tid = threadIdx.x;
    s[tid]       = iscores[(size_t)t * TT + tid];
    s[tid + 512] = iscores[(size_t)t * TT + tid + 512];
    for (int k = 2; k <= 1024; k <<= 1) {
        for (int j = k >> 1; j > 0; j >>= 1) {
            __syncthreads();
            for (int i = tid; i < 1024; i += 512) {
                int l = i ^ j;
                if (l > i) {
                    float a = s[i], b = s[l];
                    bool up = ((i & k) == 0);       // ascending overall
                    if ((a > b) == up) { s[i] = b; s[l] = a; }
                }
            }
        }
    }
    __syncthreads();
    if (tid == 0) thresh[t] = s[512];   // 512th largest
}

// ---------------------------------------------------------------------------
// Row softmax (masked entries are NEGF) -> bf16 probabilities
// ---------------------------------------------------------------------------
__global__ void softmax_k(const float* __restrict__ att, __bf16* __restrict__ p)
{
    __shared__ float sbuf[256];
    int t = blockIdx.x, tid = threadIdx.x;
    const float* rowp = att + (size_t)t * TT;
    float mx = -3.4e38f;
    for (int i = tid; i < TT; i += 256) mx = fmaxf(mx, rowp[i]);
    sbuf[tid] = mx; __syncthreads();
    for (int s = 128; s > 0; s >>= 1) {
        if (tid < s) sbuf[tid] = fmaxf(sbuf[tid], sbuf[tid + s]);
        __syncthreads();
    }
    mx = sbuf[0]; __syncthreads();
    float sum = 0.0f;
    for (int i = tid; i < TT; i += 256) sum += __expf(rowp[i] - mx);
    sum = block_reduce_sum256(sum, sbuf);
    float inv = 1.0f / sum;
    for (int i = tid; i < TT; i += 256)
        p[(size_t)t * TT + i] = (__bf16)(__expf(rowp[i] - mx) * inv);
}

// ---------------------------------------------------------------------------
// MoE gate: logits, sigmoid, grouped top-2-of-4-groups, top-4 experts -> comb
// ---------------------------------------------------------------------------
__global__ void gate_k(const float* __restrict__ h2, const float* __restrict__ gw,
                       const float* __restrict__ gb, float* __restrict__ comb)
{
    __shared__ float sbuf[256];
    __shared__ float glog[8];
    int t = blockIdx.x, tid = threadIdx.x;
    for (int e = 0; e < 8; ++e) {
        float p = 0.0f;
        for (int i = tid; i < HID; i += 256)
            p += h2[(size_t)t * HID + i] * gw[e * HID + i];
        p = block_reduce_sum256(p, sbuf);
        if (tid == 0) glog[e] = p;
    }
    __syncthreads();
    if (tid == 0) {
        float sg[8], sc[8];
        for (int e = 0; e < 8; ++e) {
            sg[e] = 1.0f / (1.0f + __expf(-glog[e]));
            sc[e] = sg[e] + gb[e];
        }
        float gs[4];
        for (int g = 0; g < 4; ++g) gs[g] = sc[2 * g] + sc[2 * g + 1];
        bool gm[4] = {false, false, false, false};
        for (int k = 0; k < 2; ++k) {
            int bi = 0; float bv = -3.4e38f;
            for (int g = 0; g < 4; ++g)
                if (!gm[g] && gs[g] > bv) { bv = gs[g]; bi = g; }
            gm[bi] = true;
        }
        bool used[8] = {false, false, false, false, false, false, false, false};
        int id[4]; float tsum = 0.0f;
        for (int k = 0; k < 4; ++k) {
            int bi = 0; float bv = -3.4e38f;
            for (int e = 0; e < 8; ++e) {
                float v = (gm[e >> 1] && !used[e]) ? sc[e] : NEGF;
                if (v > bv) { bv = v; bi = e; }
            }
            used[bi] = true; id[k] = bi; tsum += sg[bi];
        }
        float cw[8] = {0, 0, 0, 0, 0, 0, 0, 0};
        for (int k = 0; k < 4; ++k)
            cw[id[k]] = sg[id[k]] / (tsum + 1e-20f) * 2.5f;
        for (int e = 0; e < 8; ++e) comb[t * 8 + e] = cw[e];
    }
}

// ---------------------------------------------------------------------------
// silu(gate)*up (optionally scaled by routing weight) -> bf16
// gu layout [T][1024] : gate [0,512) | up [512,1024)
// ---------------------------------------------------------------------------
__global__ void silu_k(const float* __restrict__ gu, const float* __restrict__ comb,
                       int e, __bf16* __restrict__ out)
{
    int idx = blockIdx.x * blockDim.x + threadIdx.x;
    if (idx >= TT * 512) return;
    int t = idx >> 9, i = idx & 511;
    float g = gu[(size_t)t * 1024 + i];
    float u = gu[(size_t)t * 1024 + 512 + i];
    float v = g / (1.0f + __expf(-g)) * u;
    if (comb) v *= comb[t * 8 + e];
    out[idx] = (__bf16)v;
}

// ---------------------------------------------------------------------------
// Orchestration
// ---------------------------------------------------------------------------
static void cvt(hipStream_t s, const float* in, __bf16* out, int n)
{
    cvt_bf16_k<<<(n + 255) / 256, 256, 0, s>>>(in, out, n);
}

extern "C" void kernel_launch(void* const* d_in, const int* in_sizes, int n_in,
                              void* d_out, int out_size, void* d_ws, size_t ws_size,
                              hipStream_t stream)
{
    const int*   positions  = (const int*)d_in[0];
    const float* hidden     = (const float*)d_in[1];
    const float* residual   = (const float*)d_in[2];
    const float* w_input_ln = (const float*)d_in[3];
    const float* w_post_ln  = (const float*)d_in[4];
    const float* w_step1    = (const float*)d_in[5];
    const float* w_q_a_ln   = (const float*)d_in[6];
    const float* w_kv_a_ln  = (const float*)d_in[7];
    const float* idx_k_ln_w = (const float*)d_in[8];
    const float* idx_k_ln_b = (const float*)d_in[9];
    const float* w_step3    = (const float*)d_in[10];
    const float* W_UK_T     = (const float*)d_in[11];
    const float* W_UV       = (const float*)d_in[12];
    const float* w_o        = (const float*)d_in[13];
    const float* gate_w     = (const float*)d_in[14];
    const float* gate_bias  = (const float*)d_in[15];
    const float* shared_gu  = (const float*)d_in[16];
    const float* shared_dn  = (const float*)d_in[17];
    const float* exp_gu     = (const float*)d_in[18];
    const float* exp_dn     = (const float*)d_in[19];

    char* wp = (char*)d_ws;
    auto alloc = [&](size_t bytes) -> char* {
        char* r = wp;
        wp += (bytes + 255) & ~(size_t)255;
        return r;
    };

    // bf16 weight copies (no transposes needed: BN GEMM mode handles W_UK_T/W_UV)
    __bf16* w1_bf    = (__bf16*)alloc((size_t)1504 * 2048 * 2);
    __bf16* w3_bf    = (__bf16*)alloc((size_t)7168 * 768 * 2);
    __bf16* wukt_bf  = (__bf16*)alloc((size_t)16 * 128 * 512 * 2);
    __bf16* wuv_bf   = (__bf16*)alloc((size_t)16 * 512 * 128 * 2);
    __bf16* wo_bf    = (__bf16*)alloc((size_t)2048 * 2048 * 2);
    __bf16* sgu_bf   = (__bf16*)alloc((size_t)1024 * 2048 * 2);
    __bf16* sdn_bf   = (__bf16*)alloc((size_t)2048 * 512 * 2);
    __bf16* egu_bf   = (__bf16*)alloc((size_t)8 * 1024 * 2048 * 2);
    __bf16* edn_bf   = (__bf16*)alloc((size_t)8 * 2048 * 512 * 2);

    // activations
    float*  res      = (float*) alloc((size_t)TT * HID * 4);
    __bf16* h_bf     = (__bf16*)alloc((size_t)TT * HID * 2);
    float*  s1       = (float*) alloc((size_t)TT * 1504 * 4);
    __bf16* qc_bf    = (__bf16*)alloc((size_t)TT * 768 * 2);
    __bf16* kv_bf    = (__bf16*)alloc((size_t)TT * 576 * 2);
    __bf16* idxk_bf  = (__bf16*)alloc((size_t)TT * 128 * 2);
    float*  idxw     = (float*) alloc((size_t)TT * 32 * 4);
    float*  s3       = (float*) alloc((size_t)TT * 7168 * 4);
    __bf16* idxq_bf  = (__bf16*)alloc((size_t)TT * 4096 * 2);
    __bf16* qnope_bf = (__bf16*)alloc((size_t)TT * 2048 * 2);
    __bf16* mqa_bf   = (__bf16*)alloc((size_t)16 * TT * 576 * 2);
    float*  iscores  = (float*) alloc((size_t)TT * TT * 4);
    float*  thresh   = (float*) alloc((size_t)TT * 4);
    float*  att      = (float*) alloc((size_t)TT * TT * 4);
    __bf16* p_bf     = (__bf16*)alloc((size_t)TT * TT * 2);
    __bf16* oh_bf    = (__bf16*)alloc((size_t)TT * 512 * 2);
    __bf16* o2_bf    = (__bf16*)alloc((size_t)TT * 2048 * 2);
    float*  attn_out = (float*) alloc((size_t)TT * HID * 4);
    float*  h2f      = (float*) alloc((size_t)TT * HID * 4);
    __bf16* h2_bf    = (__bf16*)alloc((size_t)TT * HID * 2);
    float*  comb     = (float*) alloc((size_t)TT * 8 * 4);
    float*  gus      = (float*) alloc((size_t)TT * 1024 * 4);
    __bf16* act_bf   = (__bf16*)alloc((size_t)TT * 512 * 2);

    float* out_main = (float*)d_out;                       // [T, HID]
    float* res2_out = (float*)d_out + (size_t)TT * HID;    // [T, HID]

    // ---- 0. weight conversion ----
    cvt(stream, w_step1,   w1_bf,   1504 * 2048);
    cvt(stream, w_step3,   w3_bf,   7168 * 768);
    cvt(stream, W_UK_T,    wukt_bf, 16 * 128 * 512);
    cvt(stream, W_UV,      wuv_bf,  16 * 512 * 128);
    cvt(stream, w_o,       wo_bf,   2048 * 2048);
    cvt(stream, shared_gu, sgu_bf,  1024 * 2048);
    cvt(stream, shared_dn, sdn_bf,  2048 * 512);
    cvt(stream, exp_gu,    egu_bf,  8 * 1024 * 2048);
    cvt(stream, exp_dn,    edn_bf,  8 * 2048 * 512);

    // ---- 1. res = hidden + residual ; h = rms(res) ----
    resrms_k<<<TT, 256, 0, stream>>>(hidden, residual, w_input_ln, res, h_bf, nullptr);

    // ---- 2. s1 = h @ w_step1.T ----
    gemm<true, 0, 0>(stream, h_bf, HID, w1_bf, HID, s1, 1504, TT, 1504, HID);

    // ---- 3. norms / rope / indexer-k prep ----
    prep1_k<<<TT, 256, 0, stream>>>(s1, positions, w_q_a_ln, w_kv_a_ln,
                                    idx_k_ln_w, idx_k_ln_b,
                                    qc_bf, kv_bf, idxk_bf, idxw);

    // ---- 4. s3 = q_c @ w_step3.T ----
    gemm<true, 0, 0>(stream, qc_bf, 768, w3_bf, 768, s3, 7168, TT, 7168, 768);

    // ---- 5. idx_q rope/scale, q_nope, q_pe rope ----
    prep2_k<<<TT, 256, 0, stream>>>(s3, positions, idxq_bf, qnope_bf, mqa_bf);

    // ---- 6. ql_nope[h] = q_nope[:,h,:] @ W_UK_T[h]  (BN mode, bf16 out) ----
    for (int h = 0; h < 16; ++h)
        gemm<false, 2, 0>(stream, qnope_bf + h * 128, 2048,
                          wukt_bf + (size_t)h * 128 * 512, 512,
                          mqa_bf + (size_t)h * TT * 576, 576, TT, 512, 128);

    // ---- 7. indexer scores + top-512 threshold ----
    {
        dim3 g(TT, TT / 64);
        idx_score_k<<<g, 64, 0, stream>>>(idxq_bf, idxk_bf, idxw, iscores);
    }
    topk_thresh_k<<<TT, 512, 0, stream>>>(iscores, thresh);

    // ---- 8. per-head sparse attention ----
    const float att_scale = 0.07216878364870323f;  // (128+64)^-0.5
    for (int h = 0; h < 16; ++h) {
        gemm<true, 0, 1>(stream, mqa_bf + (size_t)h * TT * 576, 576,
                         kv_bf, 576, att, TT, TT, TT, 576,
                         att_scale, iscores, thresh);
        softmax_k<<<TT, 256, 0, stream>>>(att, p_bf);
        // o_h = p @ kv_c   (kv_bf cols [0,512), BN mode)
        gemm<false, 2, 0>(stream, p_bf, TT, kv_bf, 576, oh_bf, 512, TT, 512, TT);
        // o2[:, h*128: ] = o_h @ W_UV[h]   (BN mode)
        gemm<false, 2, 0>(stream, oh_bf, 512,
                          wuv_bf + (size_t)h * 512 * 128, 128,
                          o2_bf + h * 128, 2048, TT, 128, 512);
    }

    // ---- 9. attn_out = o2 @ w_o.T ; res2 ; h2 = rms(res2) ----
    gemm<true, 0, 0>(stream, o2_bf, 2048, wo_bf, 2048, attn_out, HID, TT, HID, HID);
    resrms_k<<<TT, 256, 0, stream>>>(attn_out, res, w_post_ln, res2_out, h2_bf, h2f);

    // ---- 10. MoE gate / routing ----
    gate_k<<<TT, 256, 0, stream>>>(h2f, gate_w, gate_bias, comb);

    // ---- 11. shared expert (writes d_out) ----
    gemm<true, 0, 0>(stream, h2_bf, HID, sgu_bf, HID, gus, 1024, TT, 1024, HID);
    silu_k<<<(TT * 512 + 255) / 256, 256, 0, stream>>>(gus, nullptr, 0, act_bf);
    gemm<true, 0, 0>(stream, act_bf, 512, sdn_bf, 512, out_main, HID, TT, HID, 512);

    // ---- 12. routed experts (accumulate into d_out) ----
    for (int e = 0; e < 8; ++e) {
        gemm<true, 0, 0>(stream, h2_bf, HID,
                         egu_bf + (size_t)e * 1024 * 2048, HID,
                         gus, 1024, TT, 1024, HID);
        silu_k<<<(TT * 512 + 255) / 256, 256, 0, stream>>>(gus, comb, e, act_bf);
        gemm<true, 1, 0>(stream, act_bf, 512,
                         edn_bf + (size_t)e * 2048 * 512, 512,
                         out_main, HID, TT, HID, 512);
    }
}
